// GPT_609885356940
// MI455X (gfx1250) — matrix operand order repaired and checked by reading, not compiled, not run on hardware
//
#include <hip/hip_runtime.h>
#include <hip/hip_bf16.h>
#include <math.h>
#include <stdint.h>

// ---------------------------------------------------------------------------
// GPT-2 forward (L=12,H=12,C=768,B=4,T=1024) for gfx1250 (MI455X, wave32).
// Matmuls via v_wmma_f32_16x16x32_f16; tiles staged with
// global_load_async_to_lds_b128 (ASYNCcnt) and double-buffered LDS.
// ---------------------------------------------------------------------------

typedef __attribute__((ext_vector_type(16))) _Float16 v16h;
typedef __attribute__((ext_vector_type(8)))  float    v8f;

union FragH { v16h v; uint4 q[2]; _Float16 h[16]; };

__device__ __forceinline__ v8f wmma_f16(v16h a, v16h b, v8f c) {
  // 8 args: (neg_a, A, neg_b, B, c_mod, C, reuse_a, reuse_b)
  return __builtin_amdgcn_wmma_f32_16x16x32_f16(false, a, false, b, (short)0, c,
                                                false, false);
}

// Async global -> LDS copy, 16B per lane.  VDST = LDS byte offset (hardware
// adds the wave's LDS_BASE), VADDR = 64-bit global address (GV mode).
__device__ __forceinline__ void async_copy_b128(uint32_t lds_off, const void* g) {
  asm volatile("global_load_async_to_lds_b128 %0, %1, off"
               :: "v"(lds_off), "v"((uint64_t)(uintptr_t)g)
               : "memory");
}
__device__ __forceinline__ void wait_async0() {
  asm volatile("s_wait_asynccnt 0x0" ::: "memory");
}

// Model constants
#define NB   4
#define NT   1024
#define NC   768
#define NH   12
#define ND   64
#define NM   (NB * NT)        // 4096 rows
#define QKVLD 2304            // 3*C

// ---------------------------------------------------------------------------
// elementwise: h = x + wpe[t]
// ---------------------------------------------------------------------------
__global__ __launch_bounds__(256)
void gpt_embed_add(const float* __restrict__ x, const float* __restrict__ wpe,
                   float* __restrict__ h) {
  int i = blockIdx.x * 256 + threadIdx.x;
  if (i < NM * NC) {
    int t = (i / NC) & (NT - 1);
    h[i] = x[i] + wpe[t * NC + (i % NC)];
  }
}

// ---------------------------------------------------------------------------
// f32 -> f16 weight conversion
// ---------------------------------------------------------------------------
__global__ __launch_bounds__(256)
void gpt_cvt_f16(const float* __restrict__ in, _Float16* __restrict__ out, int n) {
  int i = blockIdx.x * 256 + threadIdx.x;
  if (i < n) out[i] = (_Float16)in[i];
}

// ---------------------------------------------------------------------------
// LayerNorm: one wave per row (C=768 -> 24 elems/lane), f32 in, f16 out
// ---------------------------------------------------------------------------
__global__ __launch_bounds__(256)
void gpt_layernorm(const float* __restrict__ x, const float* __restrict__ w,
                   const float* __restrict__ b, _Float16* __restrict__ out) {
  int row  = blockIdx.x * 8 + (threadIdx.x >> 5);
  int lane = threadIdx.x & 31;
  const float* px = x + (size_t)row * NC;
  float s = 0.f, ss = 0.f;
  for (int c = lane; c < NC; c += 32) { float v = px[c]; s += v; ss += v * v; }
  for (int m = 1; m < 32; m <<= 1) { s += __shfl_xor(s, m); ss += __shfl_xor(ss, m); }
  float mean = s * (1.0f / NC);
  float var  = ss * (1.0f / NC) - mean * mean;
  float inv  = rsqrtf(var + 1e-5f);
  _Float16* po = out + (size_t)row * NC;
  for (int c = lane; c < NC; c += 32)
    po[c] = (_Float16)((px[c] - mean) * inv * w[c] + b[c]);
}

// ---------------------------------------------------------------------------
// Generic GEMM:  out[M,N] = A[M,K] @ W[N,K]^T + bias
//   MODE 0: write f16           (qkv)
//   MODE 1: exact GELU -> f16   (fc)
//   MODE 2: h += result (f32)   (proj / fcproj residual adds)
// Block tile 128x128, BK=32, 256 threads = 8 waves, each wave 64x32.
// Tiles prefetched into double-buffered LDS with async-to-LDS copies.
// ---------------------------------------------------------------------------
template <int MODE>
__global__ __launch_bounds__(256)
void gpt_gemm(const _Float16* __restrict__ A, const _Float16* __restrict__ W,
              const float* __restrict__ bias, int N, int K,
              float* __restrict__ hbuf, _Float16* __restrict__ out16) {
  __shared__ _Float16 As[2][128 * 40];
  __shared__ _Float16 Bs[2][128 * 40];
  const int tid  = threadIdx.x;
  const int lane = tid & 31, wv = tid >> 5;
  const int wrow = wv >> 2,  wcol = wv & 3;
  const int r = lane & 15,   g = lane >> 4;
  const int bm = blockIdx.y * 128, bn = blockIdx.x * 128;

  v8f acc[4][2];
  for (int mi = 0; mi < 4; ++mi)
    for (int ni = 0; ni < 2; ++ni)
      for (int j = 0; j < 8; ++j) acc[mi][ni][j] = 0.0f;

  // async-stage one 128x32 A tile + 128x32 W tile into LDS buffer `buf`
  auto stage = [&](int kb, int buf) {
    const int k0 = kb << 5;
    for (int i = tid; i < 512; i += 256) {
      const int row = i >> 2;
      const int cc  = (i & 3) << 3;
      async_copy_b128((uint32_t)(uintptr_t)&As[buf][row * 40 + cc],
                      &A[(size_t)(bm + row) * K + k0 + cc]);
      async_copy_b128((uint32_t)(uintptr_t)&Bs[buf][row * 40 + cc],
                      &W[(size_t)(bn + row) * K + k0 + cc]);
    }
  };

  const int ksteps = K >> 5;
  stage(0, 0);
  wait_async0();
  __syncthreads();

  for (int kb = 0; kb < ksteps; ++kb) {
    const int cur = kb & 1;
    if (kb + 1 < ksteps) stage(kb + 1, cur ^ 1);  // prefetch next tile

    FragH a[4], b[2];
    for (int mi = 0; mi < 4; ++mi) {
      const _Float16* p = &As[cur][(wrow * 64 + mi * 16 + r) * 40 + g * 8];
      a[mi].q[0] = *(const uint4*)p;        // k = g*8 + 0..7
      a[mi].q[1] = *(const uint4*)(p + 16); // k = 16 + g*8 + 0..7
    }
    for (int ni = 0; ni < 2; ++ni) {
      const _Float16* p = &Bs[cur][(wcol * 32 + ni * 16 + r) * 40 + g * 8];
      b[ni].q[0] = *(const uint4*)p;        // k = g*8 + 0..7
      b[ni].q[1] = *(const uint4*)(p + 16); // k = 16 + g*8 + 0..7
    }
    for (int mi = 0; mi < 4; ++mi)
      for (int ni = 0; ni < 2; ++ni)
        acc[mi][ni] = wmma_f16(a[mi].v, b[ni].v, acc[mi][ni]);

    wait_async0();     // this wave's prefetch copies landed in LDS
    __syncthreads();   // every wave's copies visible / reads of `cur` done
  }

  // epilogue (C/D layout: lane: n = r, row = j + 8*g)
  for (int mi = 0; mi < 4; ++mi)
    for (int ni = 0; ni < 2; ++ni) {
      const int col = bn + wcol * 32 + ni * 16 + r;
      const float bv = bias[col];
      for (int j = 0; j < 8; ++j) {
        const int row = bm + wrow * 64 + mi * 16 + j + 8 * g;
        float v = acc[mi][ni][j] + bv;
        if (MODE == 1)
          v = 0.5f * v * (1.0f + erff(v * 0.70710678118654752f));
        if (MODE == 2) {
          const size_t idx = (size_t)row * N + col;
          hbuf[idx] = hbuf[idx] + v;
        } else {
          out16[(size_t)row * N + col] = (_Float16)v;
        }
      }
    }
}

// ---------------------------------------------------------------------------
// Causal flash attention. One workgroup = (batch*head, 64-query tile),
// 128 threads = 4 waves; wave w owns q rows [qt*64 + w*16, +16).
// K tile async-staged into LDS (row-major), V transposed into LDS,
// Q kept in registers, P re-laid-out through LDS.
// ---------------------------------------------------------------------------
__global__ __launch_bounds__(128)
void gpt_attention(const _Float16* __restrict__ qkv, _Float16* __restrict__ y) {
  __shared__ _Float16 Ks[64 * 72];        // K tile: [k][d], ld 72
  __shared__ _Float16 Vt[64 * 72];        // V^T tile: [d][k], ld 72
  __shared__ _Float16 Pl[4][16 * 72];     // per-wave P: [q][k], ld 72

  const int bh = blockIdx.x;              // 0..47
  const int bb = bh / NH, hh = bh % NH;
  const int qt = blockIdx.y;              // 0..15
  const int tid = threadIdx.x, lane = tid & 31, wv = tid >> 5;
  const int r = lane & 15, g = lane >> 4;
  const size_t rowbase = (size_t)bb * NT;
  const int qrow = qt * 64 + wv * 16;

  // Q fragments (invariant over key tiles): A-layout, 16q x 64d, 2 k-steps
  FragH qa[2];
  for (int ks = 0; ks < 2; ++ks) {
    const _Float16* p =
        qkv + (rowbase + qrow + r) * QKVLD + hh * ND + ks * 32 + g * 8;
    qa[ks].q[0] = *(const uint4*)p;
    qa[ks].q[1] = *(const uint4*)(p + 16);
  }

  v8f O[4];
  for (int dt = 0; dt < 4; ++dt)
    for (int j = 0; j < 8; ++j) O[dt][j] = 0.0f;
  float mrow[8], lrow[8];
  for (int j = 0; j < 8; ++j) { mrow[j] = -1e30f; lrow[j] = 0.0f; }
  const float scale = 0.125f;  // 1/sqrt(64)

  for (int kt = 0; kt <= qt; ++kt) {
    __syncthreads();  // previous-tile Ks/Vt consumers done
    // async-stage K tile (row-major [k][d]): 512 x 16B chunks, 4 per thread
    for (int i = tid; i < 512; i += 128) {
      const int kk = i >> 3;
      const int cc = (i & 7) << 3;
      async_copy_b128((uint32_t)(uintptr_t)&Ks[kk * 72 + cc],
                      &qkv[(rowbase + kt * 64 + kk) * QKVLD + NC + hh * ND + cc]);
    }
    // build V^T tile in LDS (manual transpose through VGPRs)
    for (int i = tid; i < 64 * 64; i += 128) {
      const int kk = i >> 6, dd = i & 63;
      Vt[dd * 72 + kk] =
          qkv[(rowbase + kt * 64 + kk) * QKVLD + 2 * NC + hh * ND + dd];
    }
    wait_async0();    // K tile copies landed in LDS
    __syncthreads();  // (compiler waits DScnt for Vt stores before barrier)

    // S = Q @ K^T : B-fragments from LDS K tile (lane: row = key, d contig)
    v8f S[4];
    for (int nt = 0; nt < 4; ++nt)
      for (int j = 0; j < 8; ++j) S[nt][j] = 0.0f;
    for (int ks = 0; ks < 2; ++ks) {
      FragH kb4[4];
      for (int nt = 0; nt < 4; ++nt) {
        const _Float16* p = &Ks[(nt * 16 + r) * 72 + ks * 32 + g * 8];
        kb4[nt].q[0] = *(const uint4*)p;
        kb4[nt].q[1] = *(const uint4*)(p + 16);
      }
      for (int nt = 0; nt < 4; ++nt)
        S[nt] = wmma_f16(qa[ks].v, kb4[nt].v, S[nt]);
    }

    // scale + causal mask (only the diagonal tile) + row max
    const bool diag = (kt == qt);
    float rmax[8];
    for (int j = 0; j < 8; ++j) rmax[j] = -1e30f;
    for (int nt = 0; nt < 4; ++nt)
      for (int j = 0; j < 8; ++j) {
        float s = S[nt][j] * scale;
        if (diag) {
          const int kg = nt * 16 + r;
          const int qg = wv * 16 + j + 8 * g;
          if (kg > qg) s = -1e30f;
        }
        S[nt][j] = s;
        rmax[j] = fmaxf(rmax[j], s);
      }
    for (int j = 0; j < 8; ++j)
      for (int m = 1; m < 16; m <<= 1)
        rmax[j] = fmaxf(rmax[j], __shfl_xor(rmax[j], m));

    // online softmax update
    float alpha[8];
    for (int j = 0; j < 8; ++j) {
      const float mn = fmaxf(mrow[j], rmax[j]);
      alpha[j] = __expf(mrow[j] - mn);
      mrow[j] = mn;
    }
    float rsum[8];
    for (int j = 0; j < 8; ++j) rsum[j] = 0.0f;
    for (int nt = 0; nt < 4; ++nt)
      for (int j = 0; j < 8; ++j) {
        const float p = __expf(S[nt][j] - mrow[j]);
        rsum[j] += p;
        Pl[wv][(j + 8 * g) * 72 + nt * 16 + r] = (_Float16)p;
      }
    for (int j = 0; j < 8; ++j) {
      for (int m = 1; m < 16; m <<= 1) rsum[j] += __shfl_xor(rsum[j], m);
      lrow[j] = lrow[j] * alpha[j] + rsum[j];
    }
    for (int dt = 0; dt < 4; ++dt)
      for (int j = 0; j < 8; ++j) O[dt][j] *= alpha[j];

    // O += P @ V   (P as A-frag from per-wave LDS, V^T as B-frag from LDS)
    for (int ks2 = 0; ks2 < 2; ++ks2) {
      FragH pa;
      const _Float16* pp = &Pl[wv][r * 72 + ks2 * 32 + g * 8];
      pa.q[0] = *(const uint4*)pp;
      pa.q[1] = *(const uint4*)(pp + 16);
      for (int dt = 0; dt < 4; ++dt) {
        FragH vb;
        const _Float16* vp = &Vt[(dt * 16 + r) * 72 + ks2 * 32 + g * 8];
        vb.q[0] = *(const uint4*)vp;
        vb.q[1] = *(const uint4*)(vp + 16);
        O[dt] = wmma_f16(pa.v, vb.v, O[dt]);
      }
    }
  }

  // finalize: O /= l, scatter to y[4096 x 768]
  for (int dt = 0; dt < 4; ++dt)
    for (int j = 0; j < 8; ++j) {
      const float o = O[dt][j] / lrow[j];
      const size_t row = rowbase + qrow + j + 8 * g;
      y[row * NC + hh * ND + dt * 16 + r] = (_Float16)o;
    }
}

// ---------------------------------------------------------------------------
// lm head: out[row] = sum_c act[row,c] * head_w[c]   (one wave per row)
// ---------------------------------------------------------------------------
__global__ __launch_bounds__(256)
void gpt_head(const _Float16* __restrict__ act, const float* __restrict__ hw,
              float* __restrict__ out) {
  int row  = blockIdx.x * 8 + (threadIdx.x >> 5);
  int lane = threadIdx.x & 31;
  const _Float16* pa = act + (size_t)row * NC;
  float s = 0.f;
  for (int c = lane; c < NC; c += 32) s += (float)pa[c] * hw[c];
  for (int m = 1; m < 32; m <<= 1) s += __shfl_xor(s, m);
  if (lane == 0) out[row] = s;
}

// ---------------------------------------------------------------------------
// host launcher
// ---------------------------------------------------------------------------
extern "C" void kernel_launch(void* const* d_in, const int* in_sizes, int n_in,
                              void* d_out, int out_size, void* d_ws, size_t ws_size,
                              hipStream_t stream) {
  const float* x       = (const float*)d_in[0];
  const float* wpe     = (const float*)d_in[1];
  const float* ln1_w   = (const float*)d_in[2];
  const float* ln1_b   = (const float*)d_in[3];
  const float* attn_w  = (const float*)d_in[4];
  const float* attn_b  = (const float*)d_in[5];
  const float* proj_w  = (const float*)d_in[6];
  const float* proj_b  = (const float*)d_in[7];
  const float* ln2_w   = (const float*)d_in[8];
  const float* ln2_b   = (const float*)d_in[9];
  const float* fc_w    = (const float*)d_in[10];
  const float* fc_b    = (const float*)d_in[11];
  const float* fcproj_w= (const float*)d_in[12];
  const float* fcproj_b= (const float*)d_in[13];
  const float* lnf_w   = (const float*)d_in[14];
  const float* lnf_b   = (const float*)d_in[15];
  const float* head_w  = (const float*)d_in[16];

  char* ws = (char*)d_ws;
  size_t off = 0;
  auto carve = [&](size_t bytes) -> void* {
    void* p = ws + off;
    off += (bytes + 255) & ~(size_t)255;
    return p;
  };
  float*    h     = (float*)   carve((size_t)NM * NC * 4);       // residual f32
  _Float16* act0  = (_Float16*)carve((size_t)NM * NC * 2);       // LN outputs
  _Float16* act1  = (_Float16*)carve((size_t)NM * 4 * NC * 2);   // GELU(m)
  _Float16* qkv16 = (_Float16*)carve((size_t)NM * QKVLD * 2);
  _Float16* y16   = (_Float16*)carve((size_t)NM * NC * 2);
  _Float16* w16   = (_Float16*)carve((size_t)4 * NC * NC * 2);   // per-layer wt

  const int elems = NM * NC;
  gpt_embed_add<<<(elems + 255) / 256, 256, 0, stream>>>(x, wpe, h);

  for (int l = 0; l < 12; ++l) {
    // --- attention block ---
    gpt_layernorm<<<NM / 8, 256, 0, stream>>>(h, ln1_w + l * NC, ln1_b + l * NC, act0);

    const int nqkv = 3 * NC * NC;
    gpt_cvt_f16<<<(nqkv + 255) / 256, 256, 0, stream>>>(attn_w + (size_t)l * nqkv, w16, nqkv);
    gpt_gemm<0><<<dim3(QKVLD / 128, NM / 128), 256, 0, stream>>>(
        act0, w16, attn_b + l * QKVLD, QKVLD, NC, nullptr, qkv16);

    gpt_attention<<<dim3(NB * NH, NT / 64), 128, 0, stream>>>(qkv16, y16);

    const int nproj = NC * NC;
    gpt_cvt_f16<<<(nproj + 255) / 256, 256, 0, stream>>>(proj_w + (size_t)l * nproj, w16, nproj);
    gpt_gemm<2><<<dim3(NC / 128, NM / 128), 256, 0, stream>>>(
        y16, w16, proj_b + l * NC, NC, NC, h, nullptr);

    // --- MLP block ---
    gpt_layernorm<<<NM / 8, 256, 0, stream>>>(h, ln2_w + l * NC, ln2_b + l * NC, act0);

    const int nfc = 4 * NC * NC;
    gpt_cvt_f16<<<(nfc + 255) / 256, 256, 0, stream>>>(fc_w + (size_t)l * nfc, w16, nfc);
    gpt_gemm<1><<<dim3(4 * NC / 128, NM / 128), 256, 0, stream>>>(
        act0, w16, fc_b + l * 4 * NC, 4 * NC, NC, nullptr, act1);

    gpt_cvt_f16<<<(nfc + 255) / 256, 256, 0, stream>>>(fcproj_w + (size_t)l * nfc, w16, nfc);
    gpt_gemm<2><<<dim3(NC / 128, NM / 128), 256, 0, stream>>>(
        act1, w16, fcproj_b + l * NC, NC, 4 * NC, h, nullptr);
  }

  gpt_layernorm<<<NM / 8, 256, 0, stream>>>(h, lnf_w, lnf_b, act0);
  gpt_head<<<NM / 8, 256, 0, stream>>>(act0, head_w, (float*)d_out);
}
